// SparseLayerConv2D_59949153517679
// MI455X (gfx1250) — matrix operand order repaired and checked by reading, not compiled, not run on hardware
//
#include <hip/hip_runtime.h>

// ---------------- types ----------------
typedef unsigned int   u32;
typedef unsigned short u16;
typedef __attribute__((ext_vector_type(4)))  u32     u32x4;
typedef __attribute__((ext_vector_type(4)))  float   f32x4;
typedef __attribute__((ext_vector_type(4)))  u16     u16x4;
typedef __attribute__((ext_vector_type(16))) __bf16  v16bf;
typedef __attribute__((ext_vector_type(8)))  float   v8f;

union FragBF {              // 32 bytes: one WMMA 16-bit A/B operand
    v16bf v;
    u32x4 q[2];
};

// ---------------- static problem config ----------------
#define NB     16
#define HH     56
#define WW     56
#define CC     128
#define OH     54
#define OW     54
#define NFILT  256
#define KTOT   1152          // 3*3*128
#define KT     32            // WMMA K
#define NK     (KTOT/KT)     // 36 K-steps
#define BM     64            // rows per block (46656 / 64 = 729 exactly)
#define LDA    40            // LDS row pitch in ushorts (32 data + 8 pad; 80B, 16B aligned)
#define MTOT   (NB*OH*OW)    // 46656

__device__ __forceinline__ u16 f2bf(float f) {
    u32 u = __float_as_uint(f);
    u32 r = (u + 0x7FFFu + ((u >> 16) & 1u)) >> 16;   // round-to-nearest-even
    return (u16)r;
}

// LDS-only workgroup barrier: drain own DS ops, rendezvous. Deliberately does NOT
// drain LOADcnt, so register prefetches of the next tile stay in flight across it.
__device__ __forceinline__ void lds_barrier() {
    asm volatile(
        "s_wait_dscnt 0x0\n\t"
        "s_barrier_signal -1\n\t"
        "s_barrier_wait -1"
        ::: "memory");
}

// ---------------- prep kernels ----------------
__global__ void cvt_x_kernel(const float* __restrict__ x, u16* __restrict__ xbf, int n4) {
    int i = blockIdx.x * blockDim.x + threadIdx.x;
    if (i < n4) {
        f32x4 f = ((const f32x4*)x)[i];
        u16x4 o;
        o.x = f2bf(f.x); o.y = f2bf(f.y); o.z = f2bf(f.z); o.w = f2bf(f.w);
        ((u16x4*)xbf)[i] = o;
    }
}

__global__ void zero_w_kernel(u32x4* __restrict__ p, int n) {
    int i = blockIdx.x * blockDim.x + threadIdx.x;
    if (i < n) { u32x4 z = {0u, 0u, 0u, 0u}; p[i] = z; }
}

// Wt is the TRANSPOSED dense weight: Wt[n * KTOT + k]  (256 x 1152 bf16)
__global__ void scatter_w_kernel(const float* __restrict__ vals, const int* __restrict__ idx,
                                 u16* __restrict__ Wt, int nnz) {
    int i = blockIdx.x * blockDim.x + threadIdx.x;
    if (i < nnz) {
        int row = idx[2 * i + 0];   // k in [0,1152)
        int col = idx[2 * i + 1];   // n in [0,256)
        Wt[col * KTOT + row] = f2bf(vals[i]);
    }
}

// ---------------- GEMM helpers ----------------
struct TileRegs { u32x4 a, b0, b1, b2, b3; };
struct Frags    { FragBF a[2]; FragBF b[4]; };

__device__ __forceinline__ void load_tile_global(const u16* __restrict__ abase,
                                                 const u16* __restrict__ bbase,
                                                 int t, TileRegs& r) {
    const int kk = t * KT;
    const int p  = t >> 2;                 // filter tap 0..8 (K slice never crosses a tap)
    const int di = p / 3;
    const int dj = p - 3 * di;
    const int c0 = (t & 3) * 32;
    r.a  = *(const u32x4*)(abase + (di * WW + dj) * CC + c0);
    r.b0 = *(const u32x4*)(bbase + kk + 0);
    r.b1 = *(const u32x4*)(bbase + kk + 8);
    r.b2 = *(const u32x4*)(bbase + kk + 16);
    r.b3 = *(const u32x4*)(bbase + kk + 24);
}

__device__ __forceinline__ void store_tile_lds(u16* __restrict__ aldst,
                                               u16* __restrict__ bldst,
                                               const TileRegs& r) {
    *(u32x4*)(aldst)      = r.a;
    *(u32x4*)(bldst + 0)  = r.b0;
    *(u32x4*)(bldst + 8)  = r.b1;
    *(u32x4*)(bldst + 16) = r.b2;
    *(u32x4*)(bldst + 24) = r.b3;
}

__device__ __forceinline__ void load_frags(const u16* __restrict__ lA,
                                           const u16* __restrict__ lB,
                                           int wm, int wn, int lane, Frags& f) {
    // A fragments (ISA 16-bit A layout: VGPR0-3 hold K klo..klo+7, VGPR4-7 hold 16+klo..;
    // lanes>=16 shift klo by 8)
    const int klo = (lane >= 16) ? 8 : 0;
#pragma unroll
    for (int mf = 0; mf < 2; ++mf) {
        const int mrow = wm * 32 + mf * 16 + (lane & 15);
        f.a[mf].q[0] = *(const u32x4*)&lA[mrow * LDA + klo];
        f.a[mf].q[1] = *(const u32x4*)&lA[mrow * LDA + 16 + klo];
    }
    // B fragments (lane -> column; same K layout as A: K bko..bko+7, 16+bko..; lanes>=16 shift bko by 8)
    const int bko = (lane >= 16) ? 8 : 0;
#pragma unroll
    for (int nf = 0; nf < 4; ++nf) {
        const int ncol = wn * 64 + nf * 16 + (lane & 15);
        f.b[nf].q[0] = *(const u32x4*)&lB[ncol * LDA + bko];
        f.b[nf].q[1] = *(const u32x4*)&lB[ncol * LDA + bko + 16];
    }
}

__device__ __forceinline__ void do_wmma(const Frags& f, v8f acc[2][4]) {
#pragma unroll
    for (int mf = 0; mf < 2; ++mf)
#pragma unroll
        for (int nf = 0; nf < 4; ++nf)
            acc[mf][nf] = __builtin_amdgcn_wmma_f32_16x16x32_bf16(
                false, f.a[mf].v, false, f.b[nf].v,
                (short)0, acc[mf][nf], false, false);
}

// ---------------- WMMA GEMM: out(46656x256) = im2col(x) @ W + bias, relu ----------------
__global__ __launch_bounds__(256) void gemm_kernel(const u16* __restrict__ xbf,
                                                   const u16* __restrict__ Wt,
                                                   const float* __restrict__ bias,
                                                   float* __restrict__ out) {
    // double-buffered tiles: one (LDS-only) barrier per K-step
    __shared__ u16 ldsA[2 * BM * LDA];       // 10,240 B
    __shared__ u16 ldsB[2 * NFILT * LDA];    // 40,960 B

    const int tid  = threadIdx.x;
    const int lane = tid & 31;
    const int wave = tid >> 5;           // 0..7
    const int wm   = wave & 1;           // 2 waves along M (32 rows each)
    const int wn   = wave >> 1;          // 4 waves along N (64 cols each)

    // ---- A-tile loader coords: thread -> (row in tile, 8-elem segment of the 32-wide K slice)
    const int arow = tid >> 2;           // 0..63
    const int aseg = tid & 3;            // 0..3
    const int r    = blockIdx.x * BM + arow;          // global output row (< 46656, exact)
    const int b    = r / (OH * OW);
    const int rem  = r - b * (OH * OW);
    const int oh   = rem / OW;
    const int ow   = rem - oh * OW;
    const u16* abase = xbf + (((b * HH + oh) * WW) + ow) * CC + aseg * 8;
    const int  aoff  = arow * LDA + aseg * 8;

    // ---- B-tile loader coords: thread t owns weight row n = t (contiguous in K)
    const u16* bbase = Wt + tid * KTOT;
    const int  boff  = tid * LDA;

    v8f acc[2][4] = {};

    // prologue: tile 0 -> buffer 0
    TileRegs rg;
    load_tile_global(abase, bbase, 0, rg);
    store_tile_lds(&ldsA[aoff], &ldsB[boff], rg);

#pragma unroll 2
    for (int t = 0; t < NK - 1; ++t) {
        const int cur = t & 1;
        const int nxt = cur ^ 1;
        // prefetch next tile into registers; loads stay in flight across the barrier
        load_tile_global(abase, bbase, t + 1, rg);
        lds_barrier();     // buf[cur] stores visible; prior readers of buf[nxt] done
        Frags f;
        load_frags(&ldsA[cur * BM * LDA], &ldsB[cur * NFILT * LDA], wm, wn, lane, f);
        store_tile_lds(&ldsA[nxt * BM * LDA + aoff], &ldsB[nxt * NFILT * LDA + boff], rg);
        do_wmma(f, acc);
    }
    // final K-step
    lds_barrier();
    {
        const int cur = (NK - 1) & 1;
        Frags f;
        load_frags(&ldsA[cur * BM * LDA], &ldsB[cur * NFILT * LDA], wm, wn, lane, f);
        do_wmma(f, acc);
    }

    // ---- epilogue: D layout = VGPR v -> row v (lanes 0-15) / v+8 (lanes 16-31); lane -> col
    const int mtop = blockIdx.x * BM + wm * 32 + 8 * (lane >> 4);
#pragma unroll
    for (int nf = 0; nf < 4; ++nf) {
        const int ncol = wn * 64 + nf * 16 + (lane & 15);
        const float bb = bias[ncol];
#pragma unroll
        for (int mf = 0; mf < 2; ++mf) {
            const int m0 = mtop + mf * 16;
            float* o = out + (size_t)m0 * NFILT + ncol;
#pragma unroll
            for (int v = 0; v < 8; ++v) {
                float val = acc[mf][nf][v] + bb;
                o[(size_t)v * NFILT] = val > 0.f ? val : 0.f;
            }
        }
    }
}

// ---------------- host launcher ----------------
extern "C" void kernel_launch(void* const* d_in, const int* in_sizes, int n_in,
                              void* d_out, int out_size, void* d_ws, size_t ws_size,
                              hipStream_t stream) {
    const float* x       = (const float*)d_in[0];
    const float* kern    = (const float*)d_in[1];
    const float* bias    = (const float*)d_in[2];
    const int*   indices = (const int*)d_in[3];
    float* out = (float*)d_out;

    const int nx  = in_sizes[0];     // 6,422,528
    const int nnz = in_sizes[1];     // 58,982

    // workspace layout: [ x as bf16 | Wt (256 x 1152) bf16 ]
    u16* xbf = (u16*)d_ws;
    u16* Wt  = (u16*)((char*)d_ws + (size_t)nx * sizeof(u16));   // 12,845,056 B offset (16B aligned)

    // 1) x -> bf16
    {
        int n4 = nx / 4;
        cvt_x_kernel<<<(n4 + 255) / 256, 256, 0, stream>>>(x, xbf, n4);
    }
    // 2) zero dense weight, then scatter sparse values (transposed: Wt[n][k])
    {
        int nq = (NFILT * KTOT * (int)sizeof(u16)) / 16;   // 36,864 x 16B
        zero_w_kernel<<<(nq + 255) / 256, 256, 0, stream>>>((u32x4*)Wt, nq);
        scatter_w_kernel<<<(nnz + 255) / 256, 256, 0, stream>>>(kern, indices, Wt, nnz);
    }
    // 3) fused im2col + WMMA GEMM + bias + relu (double-buffered LDS pipeline)
    gemm_kernel<<<MTOT / BM, 256, 0, stream>>>(xbf, Wt, bias, out);
}
